// EMDLoss_38757784879723
// MI455X (gfx1250) — matrix operand order repaired and checked by reading, not compiled, not run on hardware
//
#include <hip/hip_runtime.h>
#include <hip/hip_bf16.h>

typedef __attribute__((ext_vector_type(16))) __bf16 v16bf;
typedef __attribute__((ext_vector_type(8)))  float  v8f;

// round-to-nearest-even float -> bf16 via bit manipulation (always compiles)
__device__ __forceinline__ __bf16 f2bf(float f) {
    unsigned u = __builtin_bit_cast(unsigned, f);
    u += 0x7FFFu + ((u >> 16) & 1u);
    unsigned short h = (unsigned short)(u >> 16);
    return __builtin_bit_cast(__bf16, h);
}

// One wave32 per (q,p) pair. 32x32 entropic-OT solve:
//   K = exp((sim-1)/eps)  (computed once, held as bf16 WMMA A-matrices: K and K^T)
//   100x: u = a / (K v);  v = b / (K^T u)       -> v_wmma_f32_16x16x32_bf16
//   out  = (T/32) * sum_ij u_i K_ij v_j sim_ij
//
// u/v are staged in LDS as *bf16* so the WMMA B operand is a raw 2x ds_load_b128
// bit-cast (no per-element cvt VALU in the hot loop).
__global__ __launch_bounds__(32)
void EMDLoss_sinkhorn_wmma_kernel(const float* __restrict__ sim,
                                  const int*   __restrict__ im_len,
                                  const int*   __restrict__ s_len,
                                  float*       __restrict__ out,
                                  int P) {
    __shared__ float  s_sim[1024];              // sim tile, row-major [32][32]
    __shared__ float  s_K[1024];                // K tile
    __shared__ float  s_tmp[32];                // matvec result staging (f32)
    __shared__ __align__(32) __bf16 s_vbf[32];  // current u or v vector, bf16

    const int lane = threadIdx.x;           // 0..31
    const int pair = blockIdx.x;
    const int q = pair / P;
    const int p = pair - q * P;

    // ---- load 32x32 sim tile (8 x b128 per lane, coalesced) ----
    const float4* g4 = (const float4*)(sim + (size_t)pair * 1024u);
    float4* l4 = (float4*)s_sim;
#pragma unroll
    for (int r = 0; r < 8; ++r)
        l4[r * 32 + lane] = g4[r * 32 + lane];
    __syncthreads();

    // ---- K = exp((sim-1)/eps), eps = 0.05 -> scale 20 (one exp per element) ----
#pragma unroll
    for (int i = 0; i < 32; ++i) {
        int j = i * 32 + lane;
        s_K[j] = __expf((s_sim[j] - 1.0f) * 20.0f);
    }
    __syncthreads();

    // ---- build bf16 A-matrices for K (rows 0-15 / 16-31) and K^T ----
    // 16-bit A layout (contiguous k, matches B): m = lane&15, element e -> k = e + (16 if lane>=16)
    const int m   = lane & 15;
    const int hi8 = (lane & 16) >> 1;   // 0 or 8
    v16bf aK0, aK1, aT0, aT1;
#pragma unroll
    for (int e = 0; e < 16; ++e) {
        int k = e | (hi8 << 1);
        aK0[e] = f2bf(s_K[m * 32 + k]);              // K[m][k]
        aK1[e] = f2bf(s_K[(m + 16) * 32 + k]);       // K[m+16][k]
        aT0[e] = f2bf(s_K[k * 32 + m]);              // K^T[m][k] = K[k][m]
        aT1[e] = f2bf(s_K[k * 32 + m + 16]);         // K^T[m+16][k]
    }

    // ---- closed-form marginals: w = mask + 1e-5, sum = len + 32e-5 ----
    const int  il = im_len[q];
    const int  sl = s_len[p];
    const float a = ((lane < il ? 1.0f : 0.0f) + 1e-5f) / ((float)il + 32.0f * 1e-5f);
    const float b = ((lane < sl ? 1.0f : 0.0f) + 1e-5f) / ((float)sl + 32.0f * 1e-5f);

    float u = 0.0f;
    float v = 1.0f;                                    // g0 = 0  ->  v0 = 1
    const v16bf* const Bp = (const v16bf*)(s_vbf + (lane & 16));  // lanes 0-15: k 0-15; 16-31: k 16-31
    const v8f zero = {};

#pragma unroll 1
    for (int it = 0; it < 100; ++it) {
        // ----- u = a / (K v) -----
        s_vbf[lane] = f2bf(v);                 // one ds_store_b16 per lane
        __syncthreads();
        v16bf Bv = *Bp;                        // two ds_load_b128, no cvt
        v8f d0 = __builtin_amdgcn_wmma_f32_16x16x32_bf16(false, aK0, false, Bv, (short)0, zero, false, false);
        v8f d1 = __builtin_amdgcn_wmma_f32_16x16x32_bf16(false, aK1, false, Bv, (short)0, zero, false, false);
        __syncthreads();
        if ((lane & 15) == 0) {                // D f32 layout: M = r + 8*(lane>=16), col 0 = lanes 0,16
            int h8 = (lane >> 1) & 8;
#pragma unroll
            for (int r = 0; r < 8; ++r) {
                s_tmp[h8 + r]      = d0[r];    // rows 0..15
                s_tmp[16 + h8 + r] = d1[r];    // rows 16..31
            }
        }
        __syncthreads();
        u = a * __builtin_amdgcn_rcpf(s_tmp[lane]);

        // ----- v = b / (K^T u) -----
        s_vbf[lane] = f2bf(u);
        __syncthreads();
        v16bf Bu = *Bp;
        d0 = __builtin_amdgcn_wmma_f32_16x16x32_bf16(false, aT0, false, Bu, (short)0, zero, false, false);
        d1 = __builtin_amdgcn_wmma_f32_16x16x32_bf16(false, aT1, false, Bu, (short)0, zero, false, false);
        __syncthreads();
        if ((lane & 15) == 0) {
            int h8 = (lane >> 1) & 8;
#pragma unroll
            for (int r = 0; r < 8; ++r) {
                s_tmp[h8 + r]      = d0[r];
                s_tmp[16 + h8 + r] = d1[r];
            }
        }
        __syncthreads();
        v = b * __builtin_amdgcn_rcpf(s_tmp[lane]);
        __syncthreads();
    }

    // ---- logits = (T/N1) * sum_ij u_i K_ij v_j sim_ij ; lane = row i ----
    s_tmp[lane] = v;                           // full-precision v for the epilogue
    __syncthreads();
    float partial = 0.0f;
#pragma unroll
    for (int j = 0; j < 32; ++j) {
        int idx = lane * 32 + j;
        partial += s_K[idx] * s_sim[idx] * s_tmp[j];
    }
    partial *= u;
#pragma unroll
    for (int off = 16; off > 0; off >>= 1)
        partial += __shfl_xor(partial, off);
    if (lane == 0)
        out[pair] = partial * (12.5f / 32.0f);
}

extern "C" void kernel_launch(void* const* d_in, const int* in_sizes, int n_in,
                              void* d_out, int out_size, void* d_ws, size_t ws_size,
                              hipStream_t stream) {
    (void)n_in; (void)d_ws; (void)ws_size; (void)out_size;
    const float* sim    = (const float*)d_in[0];   // [Q,P,32,32] f32
    // d_in[1] (im_set) and d_in[2] (s_seq) are unused by the reference output
    const int*   im_len = (const int*)d_in[3];     // [Q]
    const int*   s_len  = (const int*)d_in[4];     // [P]
    float*       out    = (float*)d_out;           // [Q,P]

    const int Q = in_sizes[3];
    const int P = in_sizes[4];
    dim3 grid(Q * P), block(32);
    EMDLoss_sinkhorn_wmma_kernel<<<grid, block, 0, stream>>>(sim, im_len, s_len, out, P);
}
